// recovery_net_41472204210963
// MI455X (gfx1250) — matrix operand order repaired and checked by reading, not compiled, not run on hardware
//
#include <hip/hip_runtime.h>

typedef __attribute__((ext_vector_type(16))) _Float16 v16h;
typedef __attribute__((ext_vector_type(8)))  float    v8f;

#define EPSV 1e-5f
#define NBRF 29
#define GRP  8          // 16-sample groups per wave

union AU { v16h h; unsigned u[8]; };
union BU { v16h h; _Float16 e[16]; };
union CU { v8f  v; float f[8]; };

// Wave-level LDS ordering: LDS ops from one wave execute in-order (DScnt),
// so cross-lane LDS RAW within a wave only needs a compiler barrier.
__device__ __forceinline__ void wave_sync() {
    __asm__ volatile("" ::: "memory");
    __builtin_amdgcn_wave_barrier();
    __asm__ volatile("" ::: "memory");
}

// A-fragment (16-bit A, 16x32): row = lane&15; K map per ISA 7.12.2.
__device__ __forceinline__ v16h load_a_frag(const _Float16* row, int lane, int kbase) {
    AU a;
    const unsigned* p = (const unsigned*)row;
    int kh = (lane >> 4) * 8;
#pragma unroll
    for (int v = 0; v < 8; ++v) {
        int k = kbase + ((v >> 2) * 16) + kh + ((v & 3) * 2);   // always even
        a.u[v] = p[k >> 1];
    }
    return a.h;
}

// B-fragment (16-bit B, 32x16): N = lane&15, k = 16*(v>>2) + 8*(lane>>4) + 2*(v&3) (+h).
// Folds the following-BN scale into the weights; zero-pads K and invalid N.
__device__ __forceinline__ v16h build_b_frag(const float* W, int IN, int n, bool valid,
                                             float scale, int lane, int kbase) {
    BU b;
    int kh = (lane >> 4) * 8;
#pragma unroll
    for (int v = 0; v < 8; ++v) {
        int k0 = kbase + ((v >> 2) * 16) + kh + ((v & 3) * 2);
        float w0 = (valid && (k0     < IN)) ? W[n * IN + k0]     * scale : 0.f;
        float w1 = (valid && (k0 + 1 < IN)) ? W[n * IN + k0 + 1] * scale : 0.f;
        b.e[2 * v]     = (_Float16)w0;
        b.e[2 * v + 1] = (_Float16)w1;
    }
    return b.h;
}

__global__ __launch_bounds__(256) void
recovery_net_fused(const float* __restrict__ xcon, const int* __restrict__ xcat,
                   const float* __restrict__ embt, const float* __restrict__ w1g,
                   const float* __restrict__ b1g,  const float* __restrict__ w2g,
                   const float* __restrict__ b2g,
                   const float* __restrict__ cbn_g, const float* __restrict__ cbn_b,
                   const float* __restrict__ cbn_m, const float* __restrict__ cbn_v,
                   const float* __restrict__ fc1_w, const float* __restrict__ fc1_b,
                   const float* __restrict__ fc2_w, const float* __restrict__ fc2_b,
                   const float* __restrict__ fc3_w, const float* __restrict__ fc3_b,
                   const float* __restrict__ fc4_w, const float* __restrict__ fc4_b,
                   const float* __restrict__ bn1_g, const float* __restrict__ bn1_b,
                   const float* __restrict__ bn1_m, const float* __restrict__ bn1_v,
                   const float* __restrict__ bn2_g, const float* __restrict__ bn2_b,
                   const float* __restrict__ bn2_m, const float* __restrict__ bn2_v,
                   const float* __restrict__ bn3_g, const float* __restrict__ bn3_b,
                   const float* __restrict__ bn3_m, const float* __restrict__ bn3_v,
                   float* __restrict__ out, int Btot)
{
    // ---- LDS: 145-entry embedding-MLP lookup table + per-wave staging ----
    __shared__ _Float16 s_g16[NBRF * 5];      // g[j][idx] = full mini-MLP output
    __shared__ _Float16 s_xin[8][16 * 64];    // K zero-padded staging buffers
    __shared__ _Float16 s_h1 [8][16 * 64];
    __shared__ _Float16 s_h2 [8][16 * 32];
    __shared__ _Float16 s_h3 [8][16 * 16];

    const int tid   = threadIdx.x;
    const int w     = tid >> 5;
    const int lane  = tid & 31;
    const int mrow  = lane & 15;
    const int khalf = lane >> 4;

    // ---- Phase 0a: zero staging (padding must be exact zeros) ----
    {
        unsigned* z;
        z = (unsigned*)&s_xin[0][0]; for (int i = tid; i < 8*512; i += 256) z[i] = 0u;
        z = (unsigned*)&s_h1 [0][0]; for (int i = tid; i < 8*512; i += 256) z[i] = 0u;
        z = (unsigned*)&s_h2 [0][0]; for (int i = tid; i < 8*256; i += 256) z[i] = 0u;
        z = (unsigned*)&s_h3 [0][0]; for (int i = tid; i < 8*128; i += 256) z[i] = 0u;
    }
    // ---- Phase 0b: the embedding path depends only on (j, idx): 29*5 = 145
    // possible outputs.  Precompute the whole mini-MLP per (j,idx) once. ----
    if (tid < NBRF * 5) {
        int j = tid / 5, idx = tid % 5;
        const float* e = embt + (j * 5 + idx) * 6;
        const float* W = w1g + j * 36;
        float o = b2g[j];
#pragma unroll
        for (int eo = 0; eo < 6; ++eo) {
            float hs = b1g[j * 6 + eo];
#pragma unroll
            for (int d = 0; d < 6; ++d) hs += e[d] * W[d * 6 + eo];
            o += fmaxf(hs, 0.f) * w2g[j * 6 + eo];
        }
        s_g16[tid] = (_Float16)o;
    }
    __syncthreads();

    // ---- Hoisted per-lane constants (built once, amortized over GRP groups) ----
    v16h B1[3][2], B2[2][2], B3;
    float bi1[3], bi2[2], bi3;
    {
#pragma unroll
        for (int nt = 0; nt < 3; ++nt) {
            int  n = nt * 16 + mrow;  bool val = n < 36;
            float rs = val ? rsqrtf(bn1_v[n] + EPSV) : 0.f;
            float sc = val ? rs * bn1_g[n] : 0.f;
            bi1[nt] = val ? (fc1_b[n] - bn1_m[n]) * sc + bn1_b[n] : 0.f;
            B1[nt][0] = build_b_frag(fc1_w, 36, n, val, sc, lane, 0);
            B1[nt][1] = build_b_frag(fc1_w, 36, n, val, sc, lane, 32);
        }
#pragma unroll
        for (int nt = 0; nt < 2; ++nt) {
            int  n = nt * 16 + mrow;  bool val = n < 18;
            float rs = val ? rsqrtf(bn2_v[n] + EPSV) : 0.f;
            float sc = val ? rs * bn2_g[n] : 0.f;
            bi2[nt] = val ? (fc2_b[n] - bn2_m[n]) * sc + bn2_b[n] : 0.f;
            B2[nt][0] = build_b_frag(fc2_w, 36, n, val, sc, lane, 0);
            B2[nt][1] = build_b_frag(fc2_w, 36, n, val, sc, lane, 32);
        }
        {
            int  n = mrow;  bool val = n < 9;
            float rs = val ? rsqrtf(bn3_v[n] + EPSV) : 0.f;
            float sc = val ? rs * bn3_g[n] : 0.f;
            bi3 = val ? (fc3_b[n] - bn3_m[n]) * sc + bn3_b[n] : 0.f;
            B3  = build_b_frag(fc3_w, 18, n, val, sc, lane, 0);
        }
    }
    float cscale[7], cshift[7];
#pragma unroll
    for (int c = 0; c < 7; ++c) {
        float rs = rsqrtf(cbn_v[c] + EPSV);
        cscale[c] = rs * cbn_g[c];
        cshift[c] = cbn_b[c] - cbn_m[c] * cscale[c];
    }
    float w4[9], b4;
#pragma unroll
    for (int k = 0; k < 9; ++k) w4[k] = fc4_w[khalf * 9 + k];
    b4 = fc4_b[khalf];

    _Float16* xrow = &s_xin[w][mrow * 64];
    const long long blockBase = (long long)blockIdx.x * (8 * GRP * 16);

    // ================= per-wave loop over GRP groups of 16 samples =================
    for (int g = 0; g < GRP; ++g) {
        const long long gb = blockBase + (long long)(w * GRP + g) * 16;
        long long s = gb + mrow; if (s >= Btot) s = Btot - 1;     // clamp (loads only)
        const bool live = (gb + mrow) < (long long)Btot;

        // ---- Phase 1: xcon BN + 29 table lookups -> s_xin (f16) ----
        if (lane < 16) {
#pragma unroll
            for (int c = 0; c < 7; ++c)
                xrow[c] = (_Float16)(xcon[s * 7 + c] * cscale[c] + cshift[c]);
#pragma unroll
            for (int j = 0; j < 15; ++j) {
                int idx = xcat[s * NBRF + j];
                xrow[7 + j] = s_g16[j * 5 + idx];
            }
        } else {
#pragma unroll
            for (int j = 15; j < NBRF; ++j) {
                int idx = xcat[s * NBRF + j];
                xrow[7 + j] = s_g16[j * 5 + idx];
            }
        }
        wave_sync();

        // ---- fc1 (WMMA): 36->36, BN1 folded, ReLU -> s_h1 ----
        {
            v16h a0 = load_a_frag(xrow, lane, 0);
            v16h a1 = load_a_frag(xrow, lane, 32);
#pragma unroll
            for (int nt = 0; nt < 3; ++nt) {
                float bi = bi1[nt];
                v8f c = {bi, bi, bi, bi, bi, bi, bi, bi};
                c = __builtin_amdgcn_wmma_f32_16x16x32_f16(false, a0, false, B1[nt][0], (short)0, c, false, false);
                c = __builtin_amdgcn_wmma_f32_16x16x32_f16(false, a1, false, B1[nt][1], (short)0, c, false, false);
                CU cu; cu.v = c;
                int n = nt * 16 + mrow;
                if (n < 36) {
                    int mb = khalf * 8;
#pragma unroll
                    for (int v = 0; v < 8; ++v)
                        s_h1[w][(mb + v) * 64 + n] = (_Float16)fmaxf(cu.f[v], 0.f);
                }
            }
        }
        wave_sync();

        // ---- fc2 (WMMA): 36->18, BN2 folded, ReLU -> s_h2 ----
        {
            const _Float16* hrow = &s_h1[w][mrow * 64];
            v16h a0 = load_a_frag(hrow, lane, 0);
            v16h a1 = load_a_frag(hrow, lane, 32);
#pragma unroll
            for (int nt = 0; nt < 2; ++nt) {
                float bi = bi2[nt];
                v8f c = {bi, bi, bi, bi, bi, bi, bi, bi};
                c = __builtin_amdgcn_wmma_f32_16x16x32_f16(false, a0, false, B2[nt][0], (short)0, c, false, false);
                c = __builtin_amdgcn_wmma_f32_16x16x32_f16(false, a1, false, B2[nt][1], (short)0, c, false, false);
                CU cu; cu.v = c;
                int n = nt * 16 + mrow;
                if (n < 18) {
                    int mb = khalf * 8;
#pragma unroll
                    for (int v = 0; v < 8; ++v)
                        s_h2[w][(mb + v) * 32 + n] = (_Float16)fmaxf(cu.f[v], 0.f);
                }
            }
        }
        wave_sync();

        // ---- fc3 (WMMA): 18->9 (K padded to 32 with zeros), ReLU -> s_h3 ----
        {
            v16h a0 = load_a_frag(&s_h2[w][mrow * 32], lane, 0);
            float bi = bi3;
            v8f c = {bi, bi, bi, bi, bi, bi, bi, bi};
            c = __builtin_amdgcn_wmma_f32_16x16x32_f16(false, a0, false, B3, (short)0, c, false, false);
            CU cu; cu.v = c;
            int n = mrow;
            if (n < 9) {
                int mb = khalf * 8;
#pragma unroll
                for (int v = 0; v < 8; ++v)
                    s_h3[w][(mb + v) * 16 + n] = (_Float16)fmaxf(cu.f[v], 0.f);
            }
        }
        wave_sync();

        // ---- fc4 (VALU) 9->2 + softmax; lanes (m, m+16) hold the two logits ----
        {
            const _Float16* hrow = &s_h3[w][mrow * 16];
            float logit = b4;
#pragma unroll
            for (int k = 0; k < 9; ++k) logit += (float)hrow[k] * w4[k];
            float other = __shfl_xor(logit, 16, 32);
            float mx = fmaxf(logit, other);
            float e0 = __expf(logit - mx);
            float e1 = __expf(other - mx);
            float p  = e0 / (e0 + e1);                   // P(o) on this lane
            float pp = __shfl_xor(p, 16, 32);            // partner probability
            if (live && lane < 16) {
                float2 pr = make_float2(p, pp);          // {P0, P1} for sample s
                *(float2*)(out + (long long)(gb + mrow) * 2) = pr;
            }
        }
        wave_sync();   // keep next group's s_xin writes ordered behind these reads
    }
}

extern "C" void kernel_launch(void* const* d_in, const int* in_sizes, int n_in,
                              void* d_out, int out_size, void* d_ws, size_t ws_size,
                              hipStream_t stream) {
    (void)n_in; (void)out_size; (void)d_ws; (void)ws_size;
    const int Btot   = in_sizes[0] / 7;                   // 1,048,576
    const int perBlk = 8 * GRP * 16;                      // 1024 samples / block
    const int blocks = (Btot + perBlk - 1) / perBlk;
    recovery_net_fused<<<blocks, 256, 0, stream>>>(
        (const float*)d_in[0],  (const int*)d_in[1],   (const float*)d_in[2],
        (const float*)d_in[3],  (const float*)d_in[4], (const float*)d_in[5],
        (const float*)d_in[6],
        (const float*)d_in[7],  (const float*)d_in[8], (const float*)d_in[9],  (const float*)d_in[10],
        (const float*)d_in[11], (const float*)d_in[12],
        (const float*)d_in[13], (const float*)d_in[14],
        (const float*)d_in[15], (const float*)d_in[16],
        (const float*)d_in[17], (const float*)d_in[18],
        (const float*)d_in[19], (const float*)d_in[20], (const float*)d_in[21], (const float*)d_in[22],
        (const float*)d_in[23], (const float*)d_in[24], (const float*)d_in[25], (const float*)d_in[26],
        (const float*)d_in[27], (const float*)d_in[28], (const float*)d_in[29], (const float*)d_in[30],
        (float*)d_out, Btot);
}
